// PointerAttention_30030411333686
// MI455X (gfx1250) — matrix operand order repaired and checked by reading, not compiled, not run on hardware
//
#include <hip/hip_runtime.h>

#define B_ 64
#define S_ 8192
#define U_ 128

typedef __attribute__((ext_vector_type(16))) __bf16 v16bf;
typedef __attribute__((ext_vector_type(8)))  __bf16 v8bf;
typedef __attribute__((ext_vector_type(4)))  __bf16 v4bf;
typedef __attribute__((ext_vector_type(8)))  float  v8f;
typedef __attribute__((ext_vector_type(4)))  float  v4f;

#define LOG2E 1.4426950408889634f

__device__ __forceinline__ float tanh_fast(float x) {
    // tanh(x) = 1 - 2/(exp(2x)+1); exp2 + rcp are single trans ops on CDNA5,
    // co-execute with XDL WMMA. Saturates correctly to +/-1 at |x| large.
    float e = __builtin_amdgcn_exp2f(x * (2.0f * LOG2E));
    return 1.0f - 2.0f * __builtin_amdgcn_rcpf(e + 1.0f);
}

// ---- Kernel 1a: qq[b][u] = dec_output[b] @ W1_w + W1_b + W2_b --------------
__global__ __launch_bounds__(128) void pa_q_kernel(
    const float* __restrict__ dec, const float* __restrict__ W1_w,
    const float* __restrict__ W1_b, const float* __restrict__ W2_b,
    float* __restrict__ qq)
{
    const int b = blockIdx.x, u = threadIdx.x;
    float s = W1_b[u] + W2_b[u];
    const float* d = dec + b * U_;
    #pragma unroll 8
    for (int k = 0; k < U_; ++k) s += d[k] * W1_w[k * U_ + u];
    qq[b * U_ + u] = s;
}

// ---- Kernel 1b: pack W2 (128x128 f32, K-major) into bf16 WMMA B-fragments --
// Fragment f = kc*8+nt split into two 16B chunks per lane (bank-conflict-free
// ds_load_b128): element e of lane's v16bf lives at
//   w2f[((f*2 + (e>>3))*32 + lane)*8 + (e&7)]
// with B(32x16) mapping: n = nt*16 + (lane&15), k = kc*32 + (lane>>4)*16 + e
__global__ __launch_bounds__(256) void pa_pack_kernel(
    const float* __restrict__ W2_w, __bf16* __restrict__ w2f)
{
    const int t = blockIdx.x * 256 + threadIdx.x;       // 0..1023
    const int lane = t & 31, f = t >> 5;
    const int kc = f >> 3, nt = f & 7;
    const int half = lane >> 4, nl = lane & 15;
    const int n = nt * 16 + nl;
    #pragma unroll
    for (int e = 0; e < 16; ++e) {
        int k = kc * 32 + half * 16 + e;
        w2f[(size_t)((f * 2 + (e >> 3)) * 32 + lane) * 8 + (e & 7)] =
            (__bf16)W2_w[k * U_ + n];
    }
}

// ---- Kernel 2: fused k-GEMM + tanh-attention score + clip + mask -----------
__global__ __launch_bounds__(256) void pa_gemm_kernel(
    const float* __restrict__ enc_outputs,   // [B,S,128] streaming (NT)
    const float* __restrict__ mask,          // [B,S]     streaming (NT)
    const float* __restrict__ qq,            // [B,128]   reused
    const __bf16* __restrict__ w2f,          // packed B fragments (reused)
    const float* __restrict__ V_w, const float* __restrict__ V_b,
    float* __restrict__ out_logits)          // [B,S]
{
    // A tile: padded row stride 136 bf16 = 272 B = 68 dwords -> bank rotation
    __shared__ __align__(16) __bf16 tile[128 * 136];            // 34816 B
    __shared__ __align__(16) __bf16 bfrag_lds[32 * 2 * 32 * 8]; // 32768 B

    const int b  = blockIdx.y;
    const int s0 = blockIdx.x * 128;
    const int t  = threadIdx.x;

    // --- stage W2 fragments (32 KB) global->LDS via CDNA5 async copies ------
    {
        unsigned lds_base = (unsigned)(uintptr_t)(&bfrag_lds[0]);
        unsigned long long gbase = (unsigned long long)w2f;
        #pragma unroll
        for (int i = 0; i < 8; ++i) {
            unsigned off = (unsigned)t * 128u + (unsigned)i * 16u;
            asm volatile(
                "global_load_async_to_lds_b128 %0, %1, %2 offset:0"
                :: "v"(lds_base + off), "v"(off), "s"(gbase)
                : "memory");
        }
    }

    // --- cooperative, coalesced NT load of 128x128 f32 tile -> bf16 LDS -----
    const float* src = enc_outputs + ((size_t)b * S_ + s0) * U_;
    #pragma unroll
    for (int i = 0; i < 16; ++i) {
        int idx = i * 256 + t;               // 4096 float4s
        int row = idx >> 5;
        int c4  = idx & 31;
        v4f v = __builtin_nontemporal_load(&((const v4f*)src)[row * 32 + c4]);
        v4bf p; p.x = (__bf16)v.x; p.y = (__bf16)v.y;
                p.z = (__bf16)v.z; p.w = (__bf16)v.w;
        *(v4bf*)&tile[row * 136 + c4 * 4] = p;
    }
    asm volatile("s_wait_asynccnt 0x0" ::: "memory");
    __syncthreads();

    const int wave = t >> 5, lane = t & 31;
    const int half = lane >> 4, nl = lane & 15;
    const __bf16* myrow = &tile[(wave * 16 + nl) * 136];
    const v8bf* bl = (const v8bf*)bfrag_lds;

    v8f acc[8];
    #pragma unroll
    for (int nt = 0; nt < 8; ++nt) acc[nt] = (v8f){0,0,0,0,0,0,0,0};

    #pragma unroll
    for (int kc = 0; kc < 4; ++kc) {
        // A(16x32 bf16): lane row m = nl, K = kc*32 + 16h + e
        v8bf alo = *(const v8bf*)(myrow + kc * 32 + half * 16);
        v8bf ahi = *(const v8bf*)(myrow + kc * 32 + half * 16 + 8);
        v16bf a = __builtin_shufflevector(alo, ahi,
                    0,1,2,3,4,5,6,7,8,9,10,11,12,13,14,15);
        #pragma unroll
        for (int nt = 0; nt < 8; ++nt) {
            int f = kc * 8 + nt;
            v8bf blo = bl[(f * 2 + 0) * 32 + lane];
            v8bf bhi = bl[(f * 2 + 1) * 32 + lane];
            v16bf bm = __builtin_shufflevector(blo, bhi,
                        0,1,2,3,4,5,6,7,8,9,10,11,12,13,14,15);
            acc[nt] = __builtin_amdgcn_wmma_f32_16x16x32_bf16(
                false, a, false, bm, (short)0, acc[nt], false, false);
        }
    }

    // epilogue: element (m = r + 8*half, n = nt*16 + nl)
    // score_row(m) = sum_n tanh(acc + qq[n]) * V_w[n]
    float p[8];
    #pragma unroll
    for (int r = 0; r < 8; ++r) p[r] = 0.f;
    #pragma unroll
    for (int nt = 0; nt < 8; ++nt) {
        int n = nt * 16 + nl;
        float qn = qq[b * U_ + n];
        float vn = V_w[n];
        #pragma unroll
        for (int r = 0; r < 8; ++r)
            p[r] += tanh_fast(acc[nt][r] + qn) * vn;
    }
    // reduce over the 16 lanes of each half (xor masks stay within the half)
    #pragma unroll
    for (int off = 1; off < 16; off <<= 1) {
        #pragma unroll
        for (int r = 0; r < 8; ++r)
            p[r] += __shfl_xor(p[r], off, 32);
    }
    if (nl == 0) {
        float vb = V_b[0];
        int srow = s0 + wave * 16 + half * 8;
        const float* mrow = mask + (size_t)b * S_ + srow;
        float* orow = out_logits + (size_t)b * S_ + srow;
        #pragma unroll
        for (int r = 0; r < 8; ++r) {
            float lg = 10.0f * tanh_fast(p[r] + vb);
            float mk = __builtin_nontemporal_load(&mrow[r]);
            orow[r] = lg - mk * 1e9f;
        }
    }
}

// ---- Kernel 3: per-batch softmax + argmax + gather -------------------------
__global__ __launch_bounds__(1024) void pa_softmax_kernel(
    const float* __restrict__ logits, const float* __restrict__ enc_input,
    float* __restrict__ probs, int* __restrict__ out_idx,
    float* __restrict__ out_gather)
{
    __shared__ float smax[1024];
    __shared__ int   sidx[1024];
    __shared__ float ssum[1024];
    const int b = blockIdx.x, t = threadIdx.x;
    const float* lg = logits + (size_t)b * S_;

    float v[8];
    float m = -INFINITY; int mi = 0;
    #pragma unroll
    for (int i = 0; i < 8; ++i) {
        int s = t + i * 1024;                 // increasing s order per thread
        v[i] = __builtin_nontemporal_load(&lg[s]);
        if (v[i] > m) { m = v[i]; mi = s; }   // strict > keeps first max
    }
    smax[t] = m; sidx[t] = mi;
    __syncthreads();
    for (int w = 512; w > 0; w >>= 1) {
        if (t < w) {
            float om = smax[t + w]; int oi = sidx[t + w];
            if (om > smax[t] || (om == smax[t] && oi < sidx[t])) {
                smax[t] = om; sidx[t] = oi;
            }
        }
        __syncthreads();
    }
    const float gmax = smax[0];
    const int   gidx = sidx[0];

    float s = 0.f;
    #pragma unroll
    for (int i = 0; i < 8; ++i)
        s += __builtin_amdgcn_exp2f((v[i] - gmax) * LOG2E);
    ssum[t] = s;
    __syncthreads();
    for (int w = 512; w > 0; w >>= 1) {
        if (t < w) ssum[t] += ssum[t + w];
        __syncthreads();
    }
    const float inv = 1.0f / ssum[0];

    float* pr = probs + (size_t)b * S_;
    #pragma unroll
    for (int i = 0; i < 8; ++i)
        __builtin_nontemporal_store(
            __builtin_amdgcn_exp2f((v[i] - gmax) * LOG2E) * inv,
            &pr[t + i * 1024]);

    if (t == 0) out_idx[b] = gidx;
    if (t < 16)
        out_gather[b * 16 + t] = enc_input[((size_t)b * S_ + gidx) * 16 + t];
}

// ---- host glue -------------------------------------------------------------
extern "C" void kernel_launch(void* const* d_in, const int* in_sizes, int n_in,
                              void* d_out, int out_size, void* d_ws, size_t ws_size,
                              hipStream_t stream) {
    const float* dec_output  = (const float*)d_in[0];
    const float* enc_input   = (const float*)d_in[1];
    const float* enc_outputs = (const float*)d_in[2];
    const float* mask        = (const float*)d_in[3];
    const float* W1_w = (const float*)d_in[4];
    const float* W1_b = (const float*)d_in[5];
    const float* W2_w = (const float*)d_in[6];
    const float* W2_b = (const float*)d_in[7];
    const float* V_w  = (const float*)d_in[8];
    const float* V_b  = (const float*)d_in[9];

    float* out        = (float*)d_out;
    float* out_logits = out;                                  // [B,S]
    float* out_probs  = out + (size_t)B_ * S_;                // [B,S]
    int*   out_idx    = (int*)(out + 2 * (size_t)B_ * S_);    // [B] int32 bits
    float* out_gather = out + 2 * (size_t)B_ * S_ + B_;       // [B,16]

    float*  qq  = (float*)d_ws;                                        // 64*128 f32
    __bf16* w2f = (__bf16*)((char*)d_ws + B_ * U_ * sizeof(float));    // 32 KB frags

    pa_q_kernel<<<B_, U_, 0, stream>>>(dec_output, W1_w, W1_b, W2_b, qq);
    pa_pack_kernel<<<4, 256, 0, stream>>>(W2_w, w2f);

    dim3 grid(S_ / 128, B_);
    pa_gemm_kernel<<<grid, 256, 0, stream>>>(enc_outputs, mask, qq, w2f,
                                             V_w, V_b, out_logits);
    pa_softmax_kernel<<<B_, 1024, 0, stream>>>(out_logits, enc_input,
                                               out_probs, out_idx, out_gather);
}
